// SPN_16741782519906
// MI455X (gfx1250) — matrix operand order repaired and checked by reading, not compiled, not run on hardware
//
#include <hip/hip_runtime.h>
#include <hip/hip_bf16.h>
#include <math.h>

// ---------------------------------------------------------------------------
// CSPN (spatial propagation) pipeline for MI455X (gfx1250, wave32):
//   conv3x3(19->64) -> gate-normalize -> 24 gated-stencil steps ->
//   conv3x3(64->128)+relu -> conv3x3(128->19)
//
// Convs: im2col GEMM on v_wmma_f32_16x16x32_f16 (f16 in, f32 acc). Each wave
// computes MT m-tiles against one shared B fragment; all MT A fragments are
// loaded as one clause before the WMMA burst so waits are partial and the
// XDL pipe overlaps the remaining loads.
// Propagation: gates stored f16 packed 8/pixel (one b128 load), r staged in
// LDS 16-row strips with zero halo -> branch-free 8-tap ds_load gather.
// Loop working set (~135MB) is L2-resident (192MB L2, 23.3TB/s HBM behind).
// ---------------------------------------------------------------------------

typedef __attribute__((ext_vector_type(16))) _Float16 v16h;
typedef __attribute__((ext_vector_type(8)))  _Float16 v8h;
typedef __attribute__((ext_vector_type(8)))  float    v8f;

#define BATCH 4
#define HH 128
#define WW 128
#define HW 16384            // 128*128
#define SPN 64              // SPN_DIM
#define NCLS 19
#define NPIX (BATCH * HW)   // 65536 GEMM columns
#define STRIP 16            // rows per step-kernel block

// neighbor offsets (dy, dx), same order as reference OFFSETS
__device__ __constant__ int c_dy[8] = {-1,-1,-1, 0, 0, 1, 1, 1};
__device__ __constant__ int c_dx[8] = {-1, 0, 1,-1, 1,-1, 0, 1};

// ---------------------------------------------------------------------------
// Weight repack: OIHW f32 -> (Mp x Kp) f16 row-major, K = c*9 + ky*3 + kx.
// ---------------------------------------------------------------------------
__global__ __launch_bounds__(256)
void spn_pack_weights(const float* __restrict__ Wsrc, _Float16* __restrict__ Wpk,
                      int Mp, int Mreal, int Cin, int Kp) {
    int tid = blockIdx.x * 256 + threadIdx.x;
    if (tid >= Mp * Kp) return;
    int k = tid % Kp;
    int m = tid / Kp;
    int Kreal = Cin * 9;
    float v = 0.f;
    if (m < Mreal && k < Kreal) {
        int c = k / 9, r = k % 9;
        v = Wsrc[((m * Cin + c) * 3 + r / 3) * 3 + (r % 3)];
    }
    Wpk[(size_t)m * Kp + k] = (_Float16)v;
}

// ---------------------------------------------------------------------------
// im2col pack: act (b, Cin, 128, 128) f32 -> Bm (NPIX x Kp) f16 row-major.
// ---------------------------------------------------------------------------
__global__ __launch_bounds__(256)
void spn_pack_im2col(const float* __restrict__ act, _Float16* __restrict__ Bm,
                     int Cin, int Kp) {
    long long tid = (long long)blockIdx.x * 256 + threadIdx.x;
    if (tid >= (long long)NPIX * Kp) return;
    int k = (int)(tid % Kp);
    int p = (int)(tid / Kp);
    int b   = p >> 14;
    int pix = p & (HW - 1);
    int i = pix >> 7;
    int j = pix & 127;
    int Kreal = Cin * 9;
    float v = 0.f;
    if (k < Kreal) {
        int c = k / 9, r = k % 9;
        int yy = i + r / 3 - 1;
        int xx = j + r % 3 - 1;
        if (yy >= 0 && yy < HH && xx >= 0 && xx < WW)
            v = act[((size_t)(b * Cin + c) * HH + yy) * WW + xx];
    }
    Bm[(size_t)p * Kp + k] = (_Float16)v;
}

// ---------------------------------------------------------------------------
// GEMM via WMMA. Each wave32: MT stacked 16x16 m-tiles x one 16-pixel p-tile.
// Per k-step: one clause of (2 B + 2*MT A) b128 loads, then MT WMMAs with
// partial loadcnt waits. Fragment layouts per ISA 7.12.2.
// ---------------------------------------------------------------------------
template <int MT>
__global__ __launch_bounds__(256)
void spn_gemm_wmma(const _Float16* __restrict__ A, const _Float16* __restrict__ Bm,
                   const float* __restrict__ bias, float* __restrict__ out,
                   int Mreal, int Kp, int relu) {
    const int lane = threadIdx.x & 31;
    const int wave = threadIdx.x >> 5;
    const int tile = blockIdx.x * 8 + wave;
    const int pTiles = NPIX / 16;            // 4096
    const int mg    = tile / pTiles;         // m-group (MT tiles of 16 rows)
    const int pTile = tile % pTiles;

    const int half = lane >> 4;              // 0: lanes 0-15, 1: lanes 16-31
    const int l15  = lane & 15;
    const int pBase = pTile * 16;

    const _Float16* brow = Bm + (size_t)(pBase + l15) * Kp + half * 8;
    const _Float16* arow[MT];
#pragma unroll
    for (int mt = 0; mt < MT; ++mt)
        arow[mt] = A + (size_t)((mg * MT + mt) * 16 + l15) * Kp + half * 8;

    v8f acc[MT];
#pragma unroll
    for (int mt = 0; mt < MT; ++mt) acc[mt] = v8f{};

    for (int kk = 0; kk < Kp; kk += 32) {
        // ---- load phase: one clause, distinct destination registers ----
        v8h b0 = *(const v8h*)(brow + kk);
        v8h b1 = *(const v8h*)(brow + kk + 16);
        v8h a0[MT], a1[MT];
#pragma unroll
        for (int mt = 0; mt < MT; ++mt) {
            a0[mt] = *(const v8h*)(arow[mt] + kk);
            a1[mt] = *(const v8h*)(arow[mt] + kk + 16);
        }
        v16h bv;
#pragma unroll
        for (int t = 0; t < 8; ++t) { bv[t] = b0[t]; bv[8 + t] = b1[t]; }
        // ---- compute phase: MT WMMAs against the shared B fragment ----
#pragma unroll
        for (int mt = 0; mt < MT; ++mt) {
            v16h av;
#pragma unroll
            for (int t = 0; t < 8; ++t) { av[t] = a0[mt][t]; av[8 + t] = a1[mt][t]; }
            acc[mt] = __builtin_amdgcn_wmma_f32_16x16x32_f16(
                          false, av, false, bv, (short)0, acc[mt], false, false);
        }
    }

    // Epilogue: lane -> pixel column N = l15, VGPR r -> M = r (+8 upper half)
    const int p   = pBase + l15;
    const int b   = p >> 14;
    const int pix = p & (HW - 1);
#pragma unroll
    for (int mt = 0; mt < MT; ++mt) {
#pragma unroll
        for (int r = 0; r < 8; ++r) {
            int m = (mg * MT + mt) * 16 + r + half * 8;
            if (m < Mreal) {
                float v = acc[mt][r] + bias[m];
                if (relu) v = fmaxf(v, 0.f);
                out[((size_t)b * Mreal + m) * HW + pix] = v;
            }
        }
    }
}

// ---------------------------------------------------------------------------
// Gate normalization. guidance (b, 8*64, h, w) f32.
//   N_k(i,j) = gate_k(i+dy,j+dx) / sum_k |gate_k(i+dy,j+dx)|
// Stores N as f16 packed [b][c][i][j][8] and gate_sum (f32) into D buffer.
// ---------------------------------------------------------------------------
__global__ __launch_bounds__(256)
void spn_gates(const float* __restrict__ guid, _Float16* __restrict__ Nh,
               float* __restrict__ gsum_out) {
    int idx = blockIdx.x * 256 + threadIdx.x;   // (b,c,i,j)
    int j = idx & 127;
    int i = (idx >> 7) & 127;
    int c = (idx >> 14) & 63;
    int b = idx >> 20;

    float g[8];
    float asum = 0.f;
#pragma unroll
    for (int k = 0; k < 8; ++k) {
        int yy = i + c_dy[k];
        int xx = j + c_dx[k];
        float v = 0.f;
        if (yy >= 0 && yy < HH && xx >= 0 && xx < WW)
            v = guid[((size_t)(b * 512 + k * 64 + c) * HH + yy) * WW + xx];
        g[k] = v;
        asum += fabsf(v);
    }
    float inv = 1.f / asum;   // matches reference (no guard)
    float gs = 0.f;
    v8h pack;
#pragma unroll
    for (int k = 0; k < 8; ++k) {
        float nk = g[k] * inv;
        gs += nk;
        pack[k] = (_Float16)nk;
    }
    *(v8h*)(Nh + (size_t)idx * 8) = pack;
    gsum_out[idx] = gs;
}

// ---------------------------------------------------------------------------
// Channel softmax -> sparse mask, folded step constants:
//   mask = (softmax_c(x) > 0.8) ? sign(x) : 0; Minv = 1-mask
//   D    = Minv * (1 - gate_sum) * x + mask * x
// Dbuf initially holds gate_sum; overwritten in place.
// ---------------------------------------------------------------------------
__global__ __launch_bounds__(256)
void spn_softmax_mask(const float* __restrict__ x, float* __restrict__ Dbuf,
                      float* __restrict__ Minv) {
    int tid = blockIdx.x * 256 + threadIdx.x;   // (b, pix)
    int b   = tid >> 14;
    int pix = tid & (HW - 1);
    const size_t base = (size_t)b * SPN * HW + pix;

    float xv[SPN];
    float mx = -1e30f;
#pragma unroll
    for (int c = 0; c < SPN; ++c) {
        xv[c] = x[base + (size_t)c * HW];
        mx = fmaxf(mx, xv[c]);
    }
    float se = 0.f;
#pragma unroll
    for (int c = 0; c < SPN; ++c) se += __expf(xv[c] - mx);
    float invse = 1.f / se;
#pragma unroll
    for (int c = 0; c < SPN; ++c) {
        float s = __expf(xv[c] - mx) * invse;
        float xm = xv[c];
        float msk = (s > 0.8f) ? ((xm > 0.f) ? 1.f : ((xm < 0.f) ? -1.f : 0.f)) : 0.f;
        float mi = 1.f - msk;
        size_t o = base + (size_t)c * HW;
        float gs = Dbuf[o];
        Dbuf[o] = mi * (1.f - gs) * xm + msk * xm;
        Minv[o] = mi;
    }
}

// ---------------------------------------------------------------------------
// One propagation step:  r' = D + Minv * sum_k N_k * r(i+dy_k, j+dx_k)
// Block = 16-row x 128-col strip of one (b,c) plane; 18x130 source rows
// staged in LDS with zero halo -> branch-free 8-tap gather, 8 outputs/thread.
// ---------------------------------------------------------------------------
__global__ __launch_bounds__(256)
void spn_step(const _Float16* __restrict__ Nh, const float* __restrict__ Rin,
              const float* __restrict__ Dc, const float* __restrict__ Minv,
              float* __restrict__ Rout) {
    __shared__ float ls[STRIP + 2][WW + 2];
    const int tid   = threadIdx.x;
    const int blk   = blockIdx.x;        // plane*8 + strip
    const int sp    = blk & 7;
    const int plane = blk >> 3;          // b*64 + c
    const int i0    = sp * STRIP;
    const float* rbase = Rin + (size_t)plane * HW;

    // stage rows i0-1 .. i0+STRIP (zero outside image); 9 loads per thread
#pragma unroll
    for (int t = tid; t < (STRIP + 2) * WW; t += 256) {
        int r = t >> 7, j = t & 127;
        int gi = i0 - 1 + r;
        ls[r][1 + j] = (gi >= 0 && gi < HH) ? rbase[gi * WW + j] : 0.f;
    }
    if (tid < 2 * (STRIP + 2))
        ls[tid >> 1][(tid & 1) ? (WW + 1) : 0] = 0.f;
    __syncthreads();

    const int ty = tid >> 7;             // 0 or 1
    const int j  = tid & 127;
#pragma unroll
    for (int rr = 0; rr < STRIP / 2; ++rr) {
        const int lr = rr * 2 + ty + 1;  // LDS row of output pixel
        const size_t idx = (size_t)plane * HW + (size_t)(i0 + lr - 1) * WW + j;
        v8h nv = *(const v8h*)(Nh + idx * 8);
        float s = 0.f;
#pragma unroll
        for (int k = 0; k < 8; ++k)
            s += (float)nv[k] * ls[lr + c_dy[k]][1 + j + c_dx[k]];
        Rout[idx] = Dc[idx] + Minv[idx] * s;
    }
}

// ---------------------------------------------------------------------------
// Launch pipeline
// ---------------------------------------------------------------------------
extern "C" void kernel_launch(void* const* d_in, const int* in_sizes, int n_in,
                              void* d_out, int out_size, void* d_ws, size_t ws_size,
                              hipStream_t stream) {
    const float* featureMap = (const float*)d_in[0];
    const float* guidance   = (const float*)d_in[1];
    const float* out1_w  = (const float*)d_in[2];
    const float* out1_b  = (const float*)d_in[3];
    const float* out2a_w = (const float*)d_in[4];
    const float* out2a_b = (const float*)d_in[5];
    const float* out2b_w = (const float*)d_in[6];
    const float* out2b_b = (const float*)d_in[7];
    float* out = (float*)d_out;
    char* ws = (char*)d_ws;

    // ---- workspace layout (bytes) ----
    constexpr size_t SZ_R     = (size_t)BATCH * SPN * HW * 4;      // 16 MiB
    constexpr size_t OFF_X    = 0;                                 // x / r ping
    constexpr size_t OFF_PONG = SZ_R;
    constexpr size_t OFF_D    = 2 * SZ_R;
    constexpr size_t OFF_MINV = 3 * SZ_R;
    constexpr size_t OFF_N    = 4 * SZ_R;                          // f16 gates, 64 MiB
    constexpr size_t SZ_N     = (size_t)BATCH * SPN * HW * 8 * 2;
    constexpr size_t OFF_IM2COL = OFF_N + SZ_N;                    // reused per conv
    constexpr size_t SZ_IM2COL  = (size_t)NPIX * 1152 * 2;         // largest (conv2b)
    constexpr size_t OFF_W1  = OFF_IM2COL + SZ_IM2COL;             // 64 x 192
    constexpr size_t OFF_W2A = OFF_W1 + (size_t)64 * 192 * 2;      // 128 x 576
    constexpr size_t OFF_W2B = OFF_W2A + (size_t)128 * 576 * 2;    // 32 x 1152

    float*    x    = (float*)(ws + OFF_X);
    float*    pong = (float*)(ws + OFF_PONG);
    float*    Dc   = (float*)(ws + OFF_D);
    float*    Minv = (float*)(ws + OFF_MINV);
    _Float16* Nh   = (_Float16*)(ws + OFF_N);
    _Float16* Bcol = (_Float16*)(ws + OFF_IM2COL);
    _Float16* W1p  = (_Float16*)(ws + OFF_W1);
    _Float16* W2ap = (_Float16*)(ws + OFF_W2A);
    _Float16* W2bp = (_Float16*)(ws + OFF_W2B);
    float*    hbuf = (float*)(ws + OFF_PONG);   // 32 MiB span (pong+D), dead by then

    const int pTiles = NPIX / 16;               // 4096

    // ---- pack weights to f16 GEMM layout ----
    spn_pack_weights<<<(64 * 192 + 255) / 256, 256, 0, stream>>>(out1_w, W1p, 64, 64, NCLS, 192);
    spn_pack_weights<<<(128 * 576 + 255) / 256, 256, 0, stream>>>(out2a_w, W2ap, 128, 128, SPN, 576);
    spn_pack_weights<<<(32 * 1152 + 255) / 256, 256, 0, stream>>>(out2b_w, W2bp, 32, NCLS, 2 * SPN, 1152);

    // ---- conv1: 19 -> 64  (1 m-group of MT=4) ----
    {
        long long tot = (long long)NPIX * 192;
        spn_pack_im2col<<<(unsigned)((tot + 255) / 256), 256, 0, stream>>>(featureMap, Bcol, NCLS, 192);
        spn_gemm_wmma<4><<<1 * pTiles / 8, 256, 0, stream>>>(W1p, Bcol, out1_b, x, 64, 192, 0);
    }

    // ---- gates, softmax mask, fused step constants ----
    spn_gates<<<(BATCH * SPN * HW) / 256, 256, 0, stream>>>(guidance, Nh, Dc);
    spn_softmax_mask<<<(BATCH * HW) / 256, 256, 0, stream>>>(x, Dc, Minv);

    // ---- 24 propagation steps, ping-pong (even count => result lands in x) ----
    for (int t = 0; t < 24; ++t) {
        const float* rin = (t & 1) ? pong : x;
        float* rout      = (t & 1) ? x : pong;
        spn_step<<<BATCH * SPN * (HH / STRIP), 256, 0, stream>>>(Nh, rin, Dc, Minv, rout);
    }

    // ---- conv2a: 64 -> 128, relu  (2 m-groups of MT=4) ----
    {
        long long tot = (long long)NPIX * 576;
        spn_pack_im2col<<<(unsigned)((tot + 255) / 256), 256, 0, stream>>>(x, Bcol, SPN, 576);
        spn_gemm_wmma<4><<<2 * pTiles / 8, 256, 0, stream>>>(W2ap, Bcol, out2a_b, hbuf, 128, 576, 1);
    }

    // ---- conv2b: 128 -> 19  (1 m-group of MT=2) ----
    {
        long long tot = (long long)NPIX * 1152;
        spn_pack_im2col<<<(unsigned)((tot + 255) / 256), 256, 0, stream>>>(hbuf, Bcol, 2 * SPN, 1152);
        spn_gemm_wmma<2><<<1 * pTiles / 8, 256, 0, stream>>>(W2bp, Bcol, out2b_b, out, NCLS, 1152, 0);
    }
}
